// CompressedLinear_46162308497811
// MI455X (gfx1250) — matrix operand order repaired and checked by reading, not compiled, not run on hardware
//
#include <hip/hip_runtime.h>

typedef __attribute__((ext_vector_type(16))) _Float16 v16h;
typedef __attribute__((ext_vector_type(8)))  float    v8f;
typedef __attribute__((ext_vector_type(4)))  unsigned int u32x4;
typedef __attribute__((ext_vector_type(8)))  unsigned int u32x8;

#define IN_DIM   4096
#define OUT_DIM  4096
#define M_DIM    8192   // B*S = 4*2048
#define LDA      40     // LDS row stride in f16 elems (32 + 8 pad -> conflict-free b128 reads)
#define KT       32

__device__ __forceinline__ unsigned short f32_to_f16u(float f) {
    union { _Float16 h; unsigned short u; } cv;
    cv.h = (_Float16)f;
    return cv.u;
}
__device__ __forceinline__ unsigned int pack2(float a, float b) {
    return (unsigned int)f32_to_f16u(a) | ((unsigned int)f32_to_f16u(b) << 16);
}

// Low 32 bits of a flat pointer to __shared__ == byte offset within the WG's LDS allocation.
__device__ __forceinline__ unsigned int lds_off(const void* p) {
    return (unsigned int)(unsigned long long)p;
}

// A fragment: 16x32 f16 rows [row0,row0+16) of row-major LDS tile (stride LDA).
__device__ __forceinline__ v16h load_frag_a(const unsigned short* tile, int row0, int lane) {
    const int half = lane >> 4, r = lane & 15;
    v16h a;
    uint4* ap = reinterpret_cast<uint4*>(&a);
    const unsigned short* p = tile + (row0 + r) * LDA + half * 8;
    ap[0] = *(const uint4*)(p);
    ap[1] = *(const uint4*)(p + 16);
    return a;
}
// B fragment: 32x16 f16 (KxN); lane%16 = column N = row n of the row-major [n][k] tile.
__device__ __forceinline__ v16h load_frag_b(const unsigned short* tile, int row0, int lane) {
    const int half = lane >> 4, r = lane & 15;
    v16h b;
    uint4* bp = reinterpret_cast<uint4*>(&b);
    const unsigned short* p = tile + (row0 + r) * LDA + half * 8;
    bp[0] = *(const uint4*)(p);
    bp[1] = *(const uint4*)(p + 16);
    return b;
}

// ---------------- Kernel 1: W = (centroids[indices] @ Pi) * row_norms  -> f16 in ws ----------
// 128x128 block tile, 8 waves in 2(M)x4(N), 4x2 accs/wave, double-buffered LDS (reg staged).
__global__ void __launch_bounds__(256)
dequant_gemm_w(const int* __restrict__ idx, const float* __restrict__ cent_g,
               const float* __restrict__ Pi, const float* __restrict__ row_norms,
               unsigned short* __restrict__ W) {
    __shared__ __align__(16) unsigned short ys[2][128 * LDA];  // y_hat [o][k] f16
    __shared__ __align__(16) unsigned short ps[2][128 * LDA];  // Pi^T   [n][k] f16
    __shared__ unsigned short cent16[16];

    const int tid = threadIdx.x;
    if (tid < 16) cent16[tid] = f32_to_f16u(cent_g[tid]);

    const int n0 = blockIdx.x * 128;  // IN dim (cols of W)
    const int o0 = blockIdx.y * 128;  // OUT dim (rows of W)
    const int wave = tid >> 5, lane = tid & 31;
    const int wm = wave & 1, wn = wave >> 1;

    const int yr = tid >> 3, yc = tid & 7;        // idx tile: rows yr+32i, float4-chunk yc
    const int kb = tid >> 5, nc = tid & 31;       // Pi tile: k-rows kb+8i, float4-chunk nc
    const int* idxg = idx + (size_t)(o0 + yr) * IN_DIM + yc * 4;
    const float* pig = Pi + (size_t)kb * IN_DIM + n0 + nc * 4;

    int4   iv[4];
    float4 pv[4];
    v8f acc[4][2] = {};

    // ---- prefetch tile 0 ----
    #pragma unroll
    for (int i = 0; i < 4; ++i) iv[i] = *(const int4*)(idxg + (size_t)(32 * i) * IN_DIM);
    #pragma unroll
    for (int i = 0; i < 4; ++i) pv[i] = *(const float4*)(pig + (size_t)(8 * i) * IN_DIM);
    __syncthreads();  // cent16 ready
    #pragma unroll
    for (int i = 0; i < 4; ++i) {
        unsigned int* dst = (unsigned int*)&ys[0][(yr + 32 * i) * LDA + yc * 4];
        dst[0] = (unsigned int)cent16[iv[i].x & 15] | ((unsigned int)cent16[iv[i].y & 15] << 16);
        dst[1] = (unsigned int)cent16[iv[i].z & 15] | ((unsigned int)cent16[iv[i].w & 15] << 16);
    }
    #pragma unroll
    for (int i = 0; i < 4; ++i) {
        const int nb = nc * 4, kr = kb + 8 * i;
        ps[0][(nb + 0) * LDA + kr] = f32_to_f16u(pv[i].x);
        ps[0][(nb + 1) * LDA + kr] = f32_to_f16u(pv[i].y);
        ps[0][(nb + 2) * LDA + kr] = f32_to_f16u(pv[i].z);
        ps[0][(nb + 3) * LDA + kr] = f32_to_f16u(pv[i].w);
    }
    __syncthreads();

    for (int kt = 0; kt < IN_DIM / KT; ++kt) {
        const int cur = kt & 1, nxt = cur ^ 1;
        const int k1 = (kt + 1) * KT;
        const bool pf = (k1 < IN_DIM);
        if (pf) {
            #pragma unroll
            for (int i = 0; i < 4; ++i) iv[i] = *(const int4*)(idxg + (size_t)(32 * i) * IN_DIM + k1);
            #pragma unroll
            for (int i = 0; i < 4; ++i) pv[i] = *(const float4*)(pig + (size_t)(8 * i + k1) * IN_DIM);
        }
        const unsigned short* yt = &ys[cur][0];
        const unsigned short* pt = &ps[cur][0];
        v16h a[4];
        #pragma unroll
        for (int mi = 0; mi < 4; ++mi) a[mi] = load_frag_a(yt, wm * 64 + mi * 16, lane);
        #pragma unroll
        for (int ni = 0; ni < 2; ++ni) {
            v16h bfr = load_frag_b(pt, wn * 32 + ni * 16, lane);
            #pragma unroll
            for (int mi = 0; mi < 4; ++mi)
                acc[mi][ni] = __builtin_amdgcn_wmma_f32_16x16x32_f16(
                    false, a[mi], false, bfr, (short)0, acc[mi][ni], false, false);
        }
        if (pf) {
            #pragma unroll
            for (int i = 0; i < 4; ++i) {
                unsigned int* dst = (unsigned int*)&ys[nxt][(yr + 32 * i) * LDA + yc * 4];
                dst[0] = (unsigned int)cent16[iv[i].x & 15] | ((unsigned int)cent16[iv[i].y & 15] << 16);
                dst[1] = (unsigned int)cent16[iv[i].z & 15] | ((unsigned int)cent16[iv[i].w & 15] << 16);
            }
            #pragma unroll
            for (int i = 0; i < 4; ++i) {
                const int nb = nc * 4, kr = kb + 8 * i;
                ps[nxt][(nb + 0) * LDA + kr] = f32_to_f16u(pv[i].x);
                ps[nxt][(nb + 1) * LDA + kr] = f32_to_f16u(pv[i].y);
                ps[nxt][(nb + 2) * LDA + kr] = f32_to_f16u(pv[i].z);
                ps[nxt][(nb + 3) * LDA + kr] = f32_to_f16u(pv[i].w);
            }
        }
        __syncthreads();
    }

    const int half = lane >> 4, r = lane & 15;
    #pragma unroll
    for (int mi = 0; mi < 4; ++mi) {
        const int gro = o0 + wm * 64 + mi * 16 + half * 8;
        #pragma unroll
        for (int ni = 0; ni < 2; ++ni) {
            const int gci = n0 + wn * 32 + ni * 16 + r;
            #pragma unroll
            for (int i = 0; i < 8; ++i)
                W[(size_t)(gro + i) * IN_DIM + gci] =
                    f32_to_f16u(acc[mi][ni][i] * row_norms[gro + i]);
        }
    }
}

// ---------------- Kernel 2: out = x @ W^T + bias ---------------------------------------------
// 128(M)x256(N) block tile, 8 waves 2(M)x4(N), 4x4 accs/wave (16 wmma / K-step).
// W tile: single TDM descriptor per K-step (TENSOR_LOAD_TO_LDS, TENSORcnt), hardware-padded
// into the LDA=40 LDS pitch.  x tile: reg-staged f32->f16 double buffer.
__global__ void __launch_bounds__(256)
x_gemm_out(const float* __restrict__ x, const unsigned short* __restrict__ W,
           const float* __restrict__ bias, float* __restrict__ out) {
    __shared__ __align__(16) unsigned short xs[2][128 * LDA];
    __shared__ __align__(16) unsigned short wt[2][256 * LDA];

    const int tid = threadIdx.x;
    const int n0 = blockIdx.x * 256;  // OUT dim
    const int m0 = blockIdx.y * 128;  // M dim
    const int wave = tid >> 5, lane = tid & 31;
    const int wm = wave & 1, wn = wave >> 1;

    const int xr = tid >> 3, xc = tid & 7;   // x: rows xr+32i, float4-chunk xc
    const float* xg = x + (size_t)(m0 + xr) * IN_DIM + xc * 4;

    // 2D TDM descriptor: tile 32(k, contiguous) x 256(n) f16 rows, row stride IN_DIM elems.
    // LDS pad: 4 DWORDs after every 16 DWORDs stored -> 80B row pitch == LDA(40) f16.
    const unsigned int TDIM = 1u << 24;  // huge tensor dims: disable OOB clipping
    auto issue_tdm = [&](int buf, int k) {
        const unsigned long long ga = (unsigned long long)(W + (size_t)n0 * IN_DIM + k);
        u32x4 g0;
        g0[0] = 1u;                                     // count=1, user load
        g0[1] = lds_off(&wt[buf][0]);                   // lds_addr (bytes)
        g0[2] = (unsigned int)ga;                       // global_addr[31:0]
        g0[3] = (unsigned int)(ga >> 32) | (2u << 30);  // global_addr[56:32] | type=2
        u32x8 g1;
        g1[0] = (1u << 16) | (1u << 20) | (3u << 22) | (3u << 25); // data_size=2B | pad_en | interval=16dw | amount=4dw
        g1[1] = (TDIM & 0xFFFFu) << 16;                 // tensor_dim0[15:0] (barrier addr 0)
        g1[2] = (TDIM >> 16) | ((TDIM & 0xFFFFu) << 16);// tensor_dim0[31:16] | tensor_dim1[15:0]
        g1[3] = (TDIM >> 16) | (32u << 16);             // tensor_dim1[31:16] | tile_dim0=32
        g1[4] = 256u;                                   // tile_dim1=256 | tile_dim2=0
        g1[5] = (unsigned int)IN_DIM;                   // tensor_dim0_stride[31:0]
        g1[6] = 0u;                                     // stride0[47:32] | stride1[15:0]
        g1[7] = 0u;                                     // stride1[47:16]
        asm volatile("tensor_load_to_lds %0, %1" :: "s"(g0), "s"(g1) : "memory");
    };

    float4 xv[4];
    v8f acc[4][4] = {};

    // ---- prefetch tile 0 ----
    if (wave == 0) issue_tdm(0, 0);
    #pragma unroll
    for (int i = 0; i < 4; ++i) xv[i] = *(const float4*)(xg + (size_t)(32 * i) * IN_DIM);
    #pragma unroll
    for (int i = 0; i < 4; ++i) {
        unsigned int* dst = (unsigned int*)&xs[0][(xr + 32 * i) * LDA + xc * 4];
        dst[0] = pack2(xv[i].x, xv[i].y);
        dst[1] = pack2(xv[i].z, xv[i].w);
    }
    if (wave == 0) __builtin_amdgcn_s_wait_tensorcnt(0);
    __syncthreads();

    for (int kt = 0; kt < IN_DIM / KT; ++kt) {
        const int cur = kt & 1, nxt = cur ^ 1;
        const int k1 = (kt + 1) * KT;
        const bool pf = (k1 < IN_DIM);
        if (pf) {
            if (wave == 0) issue_tdm(nxt, k1);
            #pragma unroll
            for (int i = 0; i < 4; ++i)
                xv[i] = *(const float4*)(xg + (size_t)(32 * i) * IN_DIM + k1);
        }
        const unsigned short* xt = &xs[cur][0];
        const unsigned short* wtt = &wt[cur][0];
        v16h a[4];
        #pragma unroll
        for (int mi = 0; mi < 4; ++mi) a[mi] = load_frag_a(xt, wm * 64 + mi * 16, lane);
        #pragma unroll
        for (int ni = 0; ni < 4; ++ni) {
            v16h bfr = load_frag_b(wtt, wn * 64 + ni * 16, lane);
            #pragma unroll
            for (int mi = 0; mi < 4; ++mi)
                acc[mi][ni] = __builtin_amdgcn_wmma_f32_16x16x32_f16(
                    false, a[mi], false, bfr, (short)0, acc[mi][ni], false, false);
        }
        if (pf) {
            #pragma unroll
            for (int i = 0; i < 4; ++i) {
                unsigned int* dst = (unsigned int*)&xs[nxt][(xr + 32 * i) * LDA + xc * 4];
                dst[0] = pack2(xv[i].x, xv[i].y);
                dst[1] = pack2(xv[i].z, xv[i].w);
            }
        }
        if (wave == 0) __builtin_amdgcn_s_wait_tensorcnt(0);
        __syncthreads();
    }

    const int half = lane >> 4, r = lane & 15;
    #pragma unroll
    for (int mi = 0; mi < 4; ++mi) {
        const int grm = m0 + wm * 64 + mi * 16 + half * 8;
        #pragma unroll
        for (int ni = 0; ni < 4; ++ni) {
            const int gcn = n0 + wn * 64 + ni * 16 + r;
            const float bv = bias[gcn];
            #pragma unroll
            for (int i = 0; i < 8; ++i)
                out[(size_t)(grm + i) * OUT_DIM + gcn] = acc[mi][ni][i] + bv;
        }
    }
}

extern "C" void kernel_launch(void* const* d_in, const int* in_sizes, int n_in,
                              void* d_out, int out_size, void* d_ws, size_t ws_size,
                              hipStream_t stream) {
    (void)in_sizes; (void)n_in; (void)out_size; (void)ws_size;
    const float* x    = (const float*)d_in[0];
    const int*   idx  = (const int*)d_in[1];
    const float* cen  = (const float*)d_in[2];
    const float* Pi   = (const float*)d_in[3];
    const float* rn   = (const float*)d_in[4];
    const float* bias = (const float*)d_in[5];
    float* out = (float*)d_out;
    unsigned short* W = (unsigned short*)d_ws;   // 4096*4096 f16 = 32 MB (L2-resident)

    dim3 blk(256);
    dequant_gemm_w<<<dim3(IN_DIM / 128, OUT_DIM / 128), blk, 0, stream>>>(idx, cen, Pi, rn, W);
    x_gemm_out<<<dim3(OUT_DIM / 256, M_DIM / 128), blk, 0, stream>>>(x, W, bias, out);
}
